// GraphUpdater_18056042512613
// MI455X (gfx1250) — matrix operand order repaired and checked by reading, not compiled, not run on hardware
//
#include <hip/hip_runtime.h>
#include <hip/hip_bf16.h>
#include <math.h>

// ---------------- constants ----------------
constexpr int Dc = 128;      // d_model
constexpr int Zc = 2;        // batch
constexpr int Nc = 2048;     // nodes
constexpr int Kc = 48;       // neighbors
constexpr int ZN = Zc * Nc;  // 4096
constexpr int K3 = 3 * Dc;   // 384
constexpr int Hf = 4 * Dc;   // ffn hidden = 512

constexpr int WAVES   = 6;               // messenger workgroup waves
constexpr int BLK_MSG = WAVES * 32;      // 192 threads
constexpr int NODES_PER_BLK = 2;         // 2 nodes * 48 rows = 6 tiles of 16

constexpr int FFN_WAVES = 4;
constexpr int FFN_BLK   = FFN_WAVES * 32;   // 128 threads
constexpr int FFN_ROWS  = FFN_WAVES * 16;   // 64 rows per block

constexpr size_t W_HALFS  = (size_t)Dc * K3 + 2 * (size_t)Dc * Dc;  // 81920 (W0t|W1t|W2t)
constexpr size_t H_HALFS  = (size_t)WAVES * 16 * Dc;                // 12288
constexpr size_t FW_HALFS = 2 * (size_t)Hf * Dc;                    // 131072 (w0t|w1t)

// ---------------- types ----------------
typedef __attribute__((ext_vector_type(16))) _Float16 v16h;
typedef __attribute__((ext_vector_type(8)))  _Float16 v8h;
typedef __attribute__((ext_vector_type(8)))  float    v8f;
typedef __attribute__((ext_vector_type(4)))  float    v4f;

typedef union { v16h v; v8h h[2]; } pack16;

// ---------------- device helpers ----------------
__device__ __forceinline__ float gelu_f(float x) {
  return 0.5f * x * (1.0f + erff(x * 0.70710678118654752f));
}
__device__ __forceinline__ float silu_f(float x) {
  return x / (1.0f + __expf(-x));
}
__device__ __forceinline__ v8h ld8h(const _Float16* p) {
  return *(const v8h*)p;
}
__device__ __forceinline__ v16h ld16h(const _Float16* p) {
  pack16 u;
  u.h[0] = *(const v8h*)p;
  u.h[1] = *(const v8h*)(p + 8);
  return u.v;
}
__device__ __forceinline__ v8h cvt8(const float* p) {
  v4f x0 = *(const v4f*)p;
  v4f x1 = *(const v4f*)(p + 4);
  v8h r;
#pragma unroll
  for (int i = 0; i < 4; ++i) { r[i] = (_Float16)x0[i]; r[4 + i] = (_Float16)x1[i]; }
  return r;
}

// ---------------- TDM weight staging (guarded; falls back to plain copy) ----
#if __has_builtin(__builtin_amdgcn_tensor_load_to_lds) && __has_builtin(__builtin_amdgcn_s_wait_tensorcnt)
#define USE_TDM 1
typedef unsigned int v4u __attribute__((ext_vector_type(4)));
typedef int v8i __attribute__((ext_vector_type(8)));
typedef int v4i __attribute__((ext_vector_type(4)));

// 1-D tensor DMA: nbytes (multiple of 8) from global gsrc -> LDS byte offset lds_off.
// D# per CDNA5 ISA (data_size=8B, 1 row, type=2 "image").
__device__ __forceinline__ void tdm_load_1d(const void* gsrc, unsigned lds_off,
                                            unsigned nbytes) {
  unsigned long long ga = (unsigned long long)(uintptr_t)gsrc;
  unsigned elems = nbytes >> 3;                       // 8-byte elements (<= 65535)
  v4u g0;
  g0[0] = 1u;                                         // count=1, user mode
  g0[1] = lds_off;                                    // lds_addr
  g0[2] = (unsigned)(ga & 0xFFFFFFFFu);               // global_addr[31:0]
  g0[3] = (unsigned)((ga >> 32) & 0x01FFFFFFu) | (2u << 30);  // addr[56:32] | type=2
  v8i g1;
  g1[0] = (int)(3u << 16);                            // wg_mask=0, data_size=8B
  g1[1] = (int)((elems & 0xFFFFu) << 16);             // tensor_dim0 lo16 (bits 63:48)
  g1[2] = (int)((elems >> 16) | (1u << 16));          // tensor_dim0 hi | tensor_dim1=1
  g1[3] = (int)((elems & 0xFFFFu) << 16);             // tile_dim0 (bits 127:112)
  g1[4] = 1;                                          // tile_dim1=1, tile_dim2=0
  g1[5] = (int)elems;                                 // tensor_dim0_stride lo32
  g1[6] = 0;
  g1[7] = 0;
  v4i z4 = {0, 0, 0, 0};
  v8i z8 = {0, 0, 0, 0, 0, 0, 0, 0};
  __builtin_amdgcn_tensor_load_to_lds(g0, g1, z4, z4, z8, 0);
}
#endif

// plain fallback / host-pass path
__device__ __forceinline__ void copy_lds_halfs(_Float16* dst, const _Float16* src,
                                               int nhalfs) {
  const int nvec = nhalfs / 8;
  v8h* d = (v8h*)dst;
  const v8h* s = (const v8h*)src;
  for (int i = threadIdx.x; i < nvec; i += blockDim.x) d[i] = s[i];
}

__device__ __forceinline__ void stage_weights(_Float16* dst, const _Float16* src,
                                              int nhalfs) {
#ifdef USE_TDM
  if (threadIdx.x < 32) {
    tdm_load_1d(src, 0u, (unsigned)(nhalfs * 2));
    __builtin_amdgcn_s_wait_tensorcnt(0);
  }
#else
  copy_lds_halfs(dst, src, nhalfs);
#endif
}

// ---------------- messenger 3-layer MLP core (one wave = one 16-row tile) ----
__device__ __forceinline__ void msg_mlp(
    v8f acc[8],
    const _Float16* __restrict__ viBase,
    const _Float16* __restrict__ vjBase,
    const float*    __restrict__ eRow,
    const _Float16* sW0, const _Float16* sW1, const _Float16* sW2,
    const float* __restrict__ b0, const float* __restrict__ b1,
    _Float16* hA, _Float16* hB, int lane)
{
  const int hi = lane >> 4;
  const int r  = lane & 15;
  const v8f vz = {0.f, 0.f, 0.f, 0.f, 0.f, 0.f, 0.f, 0.f};
#pragma unroll
  for (int t = 0; t < 8; ++t) acc[t] = vz;

  __builtin_prefetch(eRow, 0, 0);

  // layer 1: X(16x384) @ W0(384x128)
#pragma unroll
  for (int kk = 0; kk < 12; ++kk) {
    const int part = kk >> 2;
    const int ca   = (kk & 3) * 32 + hi * 16;
    pack16 a;
    if (part == 0)      { a.h[0] = ld8h(viBase + ca); a.h[1] = ld8h(viBase + ca + 8); }
    else if (part == 1) { a.h[0] = ld8h(vjBase + ca); a.h[1] = ld8h(vjBase + ca + 8); }
    else                { a.h[0] = cvt8(eRow + ca);   a.h[1] = cvt8(eRow + ca + 8); }
    const int ckb = kk * 32 + hi * 16;
#pragma unroll
    for (int t = 0; t < 8; ++t) {
      v16h b = ld16h(sW0 + (size_t)(t * 16 + r) * K3 + ckb);
      acc[t] = __builtin_amdgcn_wmma_f32_16x16x32_f16(
          false, a.v, false, b, (short)0, acc[t], false, false);
    }
  }
#pragma unroll
  for (int t = 0; t < 8; ++t) {
    const int n = t * 16 + r;
    const float bn = b0[n];
#pragma unroll
    for (int rr = 0; rr < 8; ++rr)
      hA[(rr + 8 * hi) * Dc + n] = (_Float16)gelu_f(acc[t][rr] + bn);
    acc[t] = vz;
  }
  __syncthreads();

  // layer 2
#pragma unroll
  for (int kk = 0; kk < 4; ++kk) {
    const int ca = kk * 32 + hi * 16;
    pack16 a;
    a.h[0] = ld8h(hA + r * Dc + ca);
    a.h[1] = ld8h(hA + r * Dc + ca + 8);
    const int ckb = kk * 32 + hi * 16;
#pragma unroll
    for (int t = 0; t < 8; ++t) {
      v16h b = ld16h(sW1 + (size_t)(t * 16 + r) * Dc + ckb);
      acc[t] = __builtin_amdgcn_wmma_f32_16x16x32_f16(
          false, a.v, false, b, (short)0, acc[t], false, false);
    }
  }
#pragma unroll
  for (int t = 0; t < 8; ++t) {
    const int n = t * 16 + r;
    const float bn = b1[n];
#pragma unroll
    for (int rr = 0; rr < 8; ++rr)
      hB[(rr + 8 * hi) * Dc + n] = (_Float16)gelu_f(acc[t][rr] + bn);
    acc[t] = vz;
  }
  __syncthreads();

  // layer 3
#pragma unroll
  for (int kk = 0; kk < 4; ++kk) {
    const int ca = kk * 32 + hi * 16;
    pack16 a;
    a.h[0] = ld8h(hB + r * Dc + ca);
    a.h[1] = ld8h(hB + r * Dc + ca + 8);
    const int ckb = kk * 32 + hi * 16;
#pragma unroll
    for (int t = 0; t < 8; ++t) {
      v16h b = ld16h(sW2 + (size_t)(t * 16 + r) * Dc + ckb);
      acc[t] = __builtin_amdgcn_wmma_f32_16x16x32_f16(
          false, a.v, false, b, (short)0, acc[t], false, false);
    }
  }
}

// ---------------- kernel 0: weight transpose + f16 cast ----------------
// src: (Kd, Nd) row-major fp32 -> dst: (Nd, Kd) row-major f16 (N-major)
__global__ void transpose_cast_kernel(const float* __restrict__ src,
                                      _Float16* __restrict__ dst, int Kd, int Nd) {
  int idx = blockIdx.x * blockDim.x + threadIdx.x;
  int total = Kd * Nd;
  if (idx >= total) return;
  int k = idx / Nd;
  int n = idx - k * Nd;
  dst[(size_t)n * Kd + k] = (_Float16)src[idx];
}

// ---------------- kernel 1: collapse_latent + FiLM ----------------
__global__ void __launch_bounds__(128) collapse_film_kernel(
    const float* __restrict__ latent, const float* __restrict__ nodes,
    const float* __restrict__ c1w, const float* __restrict__ g1g, const float* __restrict__ g1b,
    const float* __restrict__ c2w, const float* __restrict__ g2g, const float* __restrict__ g2b,
    const float* __restrict__ c3w,
    const float* __restrict__ fw0, const float* __restrict__ fb0,
    const float* __restrict__ fw1, const float* __restrict__ fb1,
    const float* __restrict__ fw2, const float* __restrict__ fb2,
    float* __restrict__ V0, _Float16* __restrict__ V0h)
{
  __shared__ float A[Dc];
  __shared__ float B[Dc];
  const int t = threadIdx.x;
  const size_t g = blockIdx.x;

  A[t] = latent[g * Dc + t];
  __syncthreads();
  {
    float acc = 0.f;
    const float* w = c1w + (size_t)t * 1024 + 7;
#pragma unroll 4
    for (int i = 0; i < Dc; ++i) acc += w[(size_t)i * 8] * A[i];
    B[t] = acc;
  }
  __syncthreads();
  {
    int gb = (t >> 4) << 4;
    float s = 0.f, s2 = 0.f;
#pragma unroll
    for (int i = 0; i < 16; ++i) { float v = B[gb + i]; s += v; s2 += v * v; }
    float mu = s * (1.f / 16.f);
    float var = s2 * (1.f / 16.f) - mu * mu;
    float y = (B[t] - mu) * rsqrtf(var + 1e-5f) * g1g[t] + g1b[t];
    float h = silu_f(y);
    __syncthreads();
    A[t] = h;
  }
  __syncthreads();
  {
    float acc = 0.f;
    const float* w = c2w + (size_t)t * 1024 + 7;
#pragma unroll 4
    for (int i = 0; i < Dc; ++i) acc += w[(size_t)i * 8] * A[i];
    B[t] = acc;
  }
  __syncthreads();
  {
    int gb = (t >> 4) << 4;
    float s = 0.f, s2 = 0.f;
#pragma unroll
    for (int i = 0; i < 16; ++i) { float v = B[gb + i]; s += v; s2 += v * v; }
    float mu = s * (1.f / 16.f);
    float var = s2 * (1.f / 16.f) - mu * mu;
    float y = (B[t] - mu) * rsqrtf(var + 1e-5f) * g2g[t] + g2b[t];
    float h = silu_f(y);
    __syncthreads();
    A[t] = h;
  }
  __syncthreads();
  {
    float acc = 0.f;
    const float* w = c3w + (size_t)t * Dc;
#pragma unroll 4
    for (int i = 0; i < Dc; ++i) acc += w[i] * A[i];
    B[t] = acc;
  }
  __syncthreads();
  {
    float acc = fb0[t];
#pragma unroll 4
    for (int d = 0; d < Dc; ++d) acc += B[d] * fw0[(size_t)d * Dc + t];
    float h = silu_f(acc);
    __syncthreads();
    A[t] = h;
  }
  __syncthreads();
  {
    float acc = fb1[t];
#pragma unroll 4
    for (int d = 0; d < Dc; ++d) acc += A[d] * fw1[(size_t)d * Dc + t];
    float h = silu_f(acc);
    __syncthreads();
    B[t] = h;
  }
  __syncthreads();
  {
    float ga = fb2[t];
    float be = fb2[Dc + t];
#pragma unroll 4
    for (int d = 0; d < Dc; ++d) {
      float hv = B[d];
      ga += hv * fw2[(size_t)d * 2 * Dc + t];
      be += hv * fw2[(size_t)d * 2 * Dc + Dc + t];
    }
    float v0 = ga * nodes[g * Dc + t] + be;
    V0[g * Dc + t]  = v0;
    V0h[g * Dc + t] = (_Float16)v0;
  }
}

// ---------------- kernel 2: node messenger + LN -> V1 ----------------
__global__ void __launch_bounds__(BLK_MSG) node_msg_kernel(
    const _Float16* __restrict__ Vh, const float* __restrict__ V0,
    const float* __restrict__ edges, const int* __restrict__ nbrs,
    const float* __restrict__ mask,
    const _Float16* __restrict__ gW,
    const float* __restrict__ b0, const float* __restrict__ b1, const float* __restrict__ b2,
    const float* __restrict__ lng, const float* __restrict__ lnb,
    float* __restrict__ V1, _Float16* __restrict__ V1h)
{
  extern __shared__ char smem[];
  _Float16* sW0 = (_Float16*)smem;
  _Float16* sW1 = sW0 + (size_t)Dc * K3;
  _Float16* sW2 = sW1 + (size_t)Dc * Dc;
  _Float16* hA  = sW2 + (size_t)Dc * Dc;
  _Float16* hB  = hA + H_HALFS;
  float* fext = (float*)(hB + H_HALFS);
  float* msum = fext;          // 256
  float* red  = fext + 256;    // 256

  stage_weights(sW0, gW, (int)W_HALFS);
  if (threadIdx.x < 256) msum[threadIdx.x] = 0.f;
  __syncthreads();

  const int tid  = threadIdx.x;
  const int lane = tid & 31;
  const int wave = tid >> 5;
  const int nl   = wave / 3;
  const int tin  = wave % 3;
  const size_t gnode = (size_t)blockIdx.x * NODES_PER_BLK + nl;
  const size_t zoff  = (gnode / Nc) * Nc;
  const int tbase = tin * 16;
  const int hi = lane >> 4;
  const int r  = lane & 15;

  const _Float16* viBase = Vh + gnode * Dc;
  const int j = nbrs[gnode * Kc + tbase + r];
  const _Float16* vjBase = Vh + (zoff + (size_t)j) * Dc;
  const float* eRow = edges + (gnode * Kc + tbase + r) * (size_t)Dc;

  v8f acc[8];
  msg_mlp(acc, viBase, vjBase, eRow, sW0, sW1, sW2, b0, b1,
          hA + (size_t)wave * 16 * Dc, hB + (size_t)wave * 16 * Dc, lane);

  float mk[8];
#pragma unroll
  for (int rr = 0; rr < 8; ++rr)
    mk[rr] = mask[gnode * Kc + tbase + rr + 8 * hi];
#pragma unroll
  for (int t = 0; t < 8; ++t) {
    const int n = t * 16 + r;
    const float bn = b2[n];
    float part = 0.f;
#pragma unroll
    for (int rr = 0; rr < 8; ++rr) part += mk[rr] * (acc[t][rr] + bn);
    atomicAdd(&msum[nl * Dc + n], part);
  }
  __syncthreads();

  // per-node: V1 = LN(V0 + msum)
  for (int nn = 0; nn < NODES_PER_BLK; ++nn) {
    const size_t gn2 = (size_t)blockIdx.x * NODES_PER_BLK + nn;
    float x = 0.f;
    if (tid < Dc) x = V0[gn2 * Dc + tid] + msum[nn * Dc + tid];
    __syncthreads();
    if (tid < Dc) { red[tid] = x; red[Dc + tid] = x * x; }
    __syncthreads();
    for (int s = 64; s > 0; s >>= 1) {
      if (tid < s) { red[tid] += red[tid + s]; red[Dc + tid] += red[Dc + tid + s]; }
      __syncthreads();
    }
    float mu = red[0] * (1.f / Dc);
    float var = red[Dc] * (1.f / Dc) - mu * mu;
    float rs = rsqrtf(var + 1e-5f);
    if (tid < Dc) {
      float v1 = (x - mu) * rs * lng[tid] + lnb[tid];
      V1[gn2 * Dc + tid]  = v1;
      V1h[gn2 * Dc + tid] = (_Float16)v1;
    }
    __syncthreads();
  }
}

// ---------------- kernel 2b: FFN (WMMA) + residual + LN -> V2 ----------------
__global__ void __launch_bounds__(FFN_BLK) ffn_kernel(
    const _Float16* __restrict__ V1h, const float* __restrict__ V1,
    const _Float16* __restrict__ gW,   // w0t[512][128] | w1t[128][512] f16
    const float* __restrict__ b0, const float* __restrict__ b1,
    const float* __restrict__ lng, const float* __restrict__ lnb,
    float* __restrict__ outV, _Float16* __restrict__ V2h)
{
  extern __shared__ char smem[];
  _Float16* w0t = (_Float16*)smem;                 // 512*128
  _Float16* w1t = w0t + (size_t)Hf * Dc;           // 128*512
  _Float16* hball = w1t + (size_t)Dc * Hf;         // FFN_WAVES * 16*512

  stage_weights(w0t, gW, (int)FW_HALFS);
  __syncthreads();

  const int tid  = threadIdx.x;
  const int lane = tid & 31;
  const int wave = tid >> 5;
  const int hi = lane >> 4;
  const int r  = lane & 15;
  const int rowBase = blockIdx.x * FFN_ROWS + wave * 16;
  _Float16* hb = hball + (size_t)wave * 16 * Hf;

  const _Float16* arow = V1h + (size_t)(rowBase + r) * Dc;
  pack16 a4[4];
#pragma unroll
  for (int kk = 0; kk < 4; ++kk) {
    const int ca = kk * 32 + hi * 16;
    a4[kk].h[0] = ld8h(arow + ca);
    a4[kk].h[1] = ld8h(arow + ca + 8);
  }
  const v8f vz = {0.f, 0.f, 0.f, 0.f, 0.f, 0.f, 0.f, 0.f};

  // GEMM1: (16x128) @ (128x512), N processed in 4 chunks of 128
  for (int c = 0; c < 4; ++c) {
    v8f acc[8];
#pragma unroll
    for (int t = 0; t < 8; ++t) acc[t] = vz;
#pragma unroll
    for (int kk = 0; kk < 4; ++kk) {
      const int ckb = kk * 32 + hi * 16;
#pragma unroll
      for (int t = 0; t < 8; ++t) {
        v16h b = ld16h(w0t + (size_t)(c * 128 + t * 16 + r) * Dc + ckb);
        acc[t] = __builtin_amdgcn_wmma_f32_16x16x32_f16(
            false, a4[kk].v, false, b, (short)0, acc[t], false, false);
      }
    }
#pragma unroll
    for (int t = 0; t < 8; ++t) {
      const int n = c * 128 + t * 16 + r;
      const float bn = b0[n];
#pragma unroll
      for (int rr = 0; rr < 8; ++rr)
        hb[(rr + 8 * hi) * Hf + n] = (_Float16)gelu_f(acc[t][rr] + bn);
    }
  }
  __syncthreads();

  // GEMM2: (16x512) @ (512x128)
  v8f acc[8];
#pragma unroll
  for (int t = 0; t < 8; ++t) acc[t] = vz;
#pragma unroll
  for (int kk = 0; kk < 16; ++kk) {
    const int ca = kk * 32 + hi * 16;
    pack16 a;
    a.h[0] = ld8h(hb + r * Hf + ca);
    a.h[1] = ld8h(hb + r * Hf + ca + 8);
    const int ckb = kk * 32 + hi * 16;
#pragma unroll
    for (int t = 0; t < 8; ++t) {
      v16h b = ld16h(w1t + (size_t)(t * 16 + r) * Hf + ckb);
      acc[t] = __builtin_amdgcn_wmma_f32_16x16x32_f16(
          false, a.v, false, b, (short)0, acc[t], false, false);
    }
  }

  // epilogue: x2 = V1 + ffn -> row buffer (reuse hb region as f32)
  float* rbw = (float*)hb;
#pragma unroll
  for (int t = 0; t < 8; ++t) {
    const int n = t * 16 + r;
    const float bn = b1[n];
#pragma unroll
    for (int rr = 0; rr < 8; ++rr) {
      const int m = rr + 8 * hi;
      float x2 = V1[(size_t)(rowBase + m) * Dc + n] + acc[t][rr] + bn;
      rbw[m * Dc + n] = x2;
    }
  }
  __syncthreads();

  // per-row LayerNorm
  for (int m = 0; m < 16; ++m) {
    const float* rp = rbw + m * Dc;
    const int c0 = lane * 4;
    v4f v = *(const v4f*)(rp + c0);
    float s = v[0] + v[1] + v[2] + v[3];
    float s2 = v[0] * v[0] + v[1] * v[1] + v[2] * v[2] + v[3] * v[3];
#pragma unroll
    for (int off = 16; off > 0; off >>= 1) {
      s  += __shfl_xor(s, off, 32);
      s2 += __shfl_xor(s2, off, 32);
    }
    float mu = s * (1.f / Dc);
    float var = s2 * (1.f / Dc) - mu * mu;
    float rs = rsqrtf(var + 1e-5f);
    size_t grow = (size_t)(rowBase + m) * Dc;
#pragma unroll
    for (int i = 0; i < 4; ++i) {
      int c = c0 + i;
      float o = (v[i] - mu) * rs * lng[c] + lnb[c];
      outV[grow + c] = o;
      V2h[grow + c] = (_Float16)o;
    }
  }
}

// ---------------- kernel 3: edge messenger + LN ----------------
__global__ void __launch_bounds__(BLK_MSG) edge_msg_kernel(
    const _Float16* __restrict__ Vh, const float* __restrict__ edges,
    const int* __restrict__ nbrs, const float* __restrict__ mask,
    const _Float16* __restrict__ gW,
    const float* __restrict__ b0, const float* __restrict__ b1, const float* __restrict__ b2,
    const float* __restrict__ lng, const float* __restrict__ lnb,
    float* __restrict__ outE)
{
  extern __shared__ char smem[];
  _Float16* sW0 = (_Float16*)smem;
  _Float16* sW1 = sW0 + (size_t)Dc * K3;
  _Float16* sW2 = sW1 + (size_t)Dc * Dc;
  _Float16* hA  = sW2 + (size_t)Dc * Dc;
  _Float16* hB  = hA + H_HALFS;
  float* rb = (float*)(hB + H_HALFS);

  stage_weights(sW0, gW, (int)W_HALFS);
  __syncthreads();

  const int tid  = threadIdx.x;
  const int lane = tid & 31;
  const int wave = tid >> 5;
  const int nl   = wave / 3;
  const int tin  = wave % 3;
  const size_t gnode = (size_t)blockIdx.x * NODES_PER_BLK + nl;
  const size_t zoff  = (gnode / Nc) * Nc;
  const int tbase = tin * 16;
  const int hi = lane >> 4;
  const int r  = lane & 15;

  const _Float16* viBase = Vh + gnode * Dc;
  const int j = nbrs[gnode * Kc + tbase + r];
  const _Float16* vjBase = Vh + (zoff + (size_t)j) * Dc;
  const float* eRow = edges + (gnode * Kc + tbase + r) * (size_t)Dc;

  v8f acc[8];
  msg_mlp(acc, viBase, vjBase, eRow, sW0, sW1, sW2, b0, b1,
          hA + (size_t)wave * 16 * Dc, hB + (size_t)wave * 16 * Dc, lane);

  float* rbw = rb + (size_t)wave * 16 * Dc;
  float mk[8];
#pragma unroll
  for (int rr = 0; rr < 8; ++rr)
    mk[rr] = mask[gnode * Kc + tbase + rr + 8 * hi];
#pragma unroll
  for (int t = 0; t < 8; ++t) {
    const int n = t * 16 + r;
    const float bn = b2[n];
#pragma unroll
    for (int rr = 0; rr < 8; ++rr) {
      const int m = rr + 8 * hi;
      float ev = edges[(gnode * Kc + tbase + m) * (size_t)Dc + n];
      rbw[m * Dc + n] = ev + mk[rr] * (acc[t][rr] + bn);
    }
  }
  __syncthreads();

  for (int m = 0; m < 16; ++m) {
    const float* rp = rbw + m * Dc;
    const int c0 = lane * 4;
    v4f v = *(const v4f*)(rp + c0);
    float s = v[0] + v[1] + v[2] + v[3];
    float s2 = v[0] * v[0] + v[1] * v[1] + v[2] * v[2] + v[3] * v[3];
#pragma unroll
    for (int off = 16; off > 0; off >>= 1) {
      s  += __shfl_xor(s, off, 32);
      s2 += __shfl_xor(s2, off, 32);
    }
    float mu = s * (1.f / Dc);
    float var = s2 * (1.f / Dc) - mu * mu;
    float rs = rsqrtf(var + 1e-5f);
    size_t gr = (gnode * Kc + tbase + m) * (size_t)Dc;
#pragma unroll
    for (int i = 0; i < 4; ++i) {
      int c = c0 + i;
      outE[gr + c] = (v[i] - mu) * rs * lng[c] + lnb[c];
    }
  }
}

// ---------------- host launcher ----------------
extern "C" void kernel_launch(void* const* d_in, const int* in_sizes, int n_in,
                              void* d_out, int out_size, void* d_ws, size_t ws_size,
                              hipStream_t stream) {
  (void)in_sizes; (void)n_in; (void)out_size; (void)ws_size;

  const float* latent = (const float*)d_in[0];
  const float* nodes  = (const float*)d_in[1];
  const float* edges  = (const float*)d_in[2];
  const int*   nbrs   = (const int*)d_in[3];
  const float* nmask  = (const float*)d_in[4];
  const float* c1w = (const float*)d_in[5];
  const float* g1g = (const float*)d_in[6];
  const float* g1b = (const float*)d_in[7];
  const float* c2w = (const float*)d_in[8];
  const float* g2g = (const float*)d_in[9];
  const float* g2b = (const float*)d_in[10];
  const float* c3w = (const float*)d_in[11];
  const float* fw0 = (const float*)d_in[12];
  const float* fb0 = (const float*)d_in[13];
  const float* fw1 = (const float*)d_in[14];
  const float* fb1 = (const float*)d_in[15];
  const float* fw2 = (const float*)d_in[16];
  const float* fb2 = (const float*)d_in[17];
  const float* nmw0 = (const float*)d_in[18];
  const float* nmb0 = (const float*)d_in[19];
  const float* nmw1 = (const float*)d_in[20];
  const float* nmb1 = (const float*)d_in[21];
  const float* nmw2 = (const float*)d_in[22];
  const float* nmb2 = (const float*)d_in[23];
  const float* nmlng = (const float*)d_in[24];
  const float* nmlnb = (const float*)d_in[25];
  const float* ffw0 = (const float*)d_in[26];
  const float* ffb0 = (const float*)d_in[27];
  const float* ffw1 = (const float*)d_in[28];
  const float* ffb1 = (const float*)d_in[29];
  const float* fflng = (const float*)d_in[30];
  const float* fflnb = (const float*)d_in[31];
  const float* emw0 = (const float*)d_in[32];
  const float* emb0 = (const float*)d_in[33];
  const float* emw1 = (const float*)d_in[34];
  const float* emb1 = (const float*)d_in[35];
  const float* emw2 = (const float*)d_in[36];
  const float* emb2 = (const float*)d_in[37];
  const float* emlng = (const float*)d_in[38];
  const float* emlnb = (const float*)d_in[39];

  float* outV = (float*)d_out;
  float* outE = outV + (size_t)ZN * Dc;

  // workspace carve
  char* wsp = (char*)d_ws;
  auto take = [&](size_t bytes) -> char* {
    char* p = wsp;
    wsp += (bytes + 255) & ~(size_t)255;
    return p;
  };
  _Float16* nmW = (_Float16*)take(W_HALFS * 2);
  _Float16* emW = (_Float16*)take(W_HALFS * 2);
  _Float16* ffW = (_Float16*)take(FW_HALFS * 2);
  float*    V0  = (float*)take((size_t)ZN * Dc * 4);
  _Float16* V0h = (_Float16*)take((size_t)ZN * Dc * 2);
  float*    V1  = (float*)take((size_t)ZN * Dc * 4);
  _Float16* V1h = (_Float16*)take((size_t)ZN * Dc * 2);
  _Float16* V2h = (_Float16*)take((size_t)ZN * Dc * 2);

  // kernel 0: weight transposes to f16 N-major
  {
    int t0 = K3 * Dc, t1 = Dc * Dc, t2 = Dc * Hf;
    transpose_cast_kernel<<<(t0 + 255) / 256, 256, 0, stream>>>(nmw0, nmW, K3, Dc);
    transpose_cast_kernel<<<(t1 + 255) / 256, 256, 0, stream>>>(nmw1, nmW + (size_t)Dc * K3, Dc, Dc);
    transpose_cast_kernel<<<(t1 + 255) / 256, 256, 0, stream>>>(nmw2, nmW + (size_t)Dc * K3 + (size_t)Dc * Dc, Dc, Dc);
    transpose_cast_kernel<<<(t0 + 255) / 256, 256, 0, stream>>>(emw0, emW, K3, Dc);
    transpose_cast_kernel<<<(t1 + 255) / 256, 256, 0, stream>>>(emw1, emW + (size_t)Dc * K3, Dc, Dc);
    transpose_cast_kernel<<<(t1 + 255) / 256, 256, 0, stream>>>(emw2, emW + (size_t)Dc * K3 + (size_t)Dc * Dc, Dc, Dc);
    transpose_cast_kernel<<<(t2 + 255) / 256, 256, 0, stream>>>(ffw0, ffW, Dc, Hf);
    transpose_cast_kernel<<<(t2 + 255) / 256, 256, 0, stream>>>(ffw1, ffW + (size_t)Hf * Dc, Hf, Dc);
  }

  // kernel 1: collapse + FiLM -> V0, V0h
  collapse_film_kernel<<<ZN, 128, 0, stream>>>(
      latent, nodes, c1w, g1g, g1b, c2w, g2g, g2b, c3w,
      fw0, fb0, fw1, fb1, fw2, fb2, V0, V0h);

  // kernel 2: node messenger + LN -> V1
  size_t smem_node = (W_HALFS + 2 * H_HALFS) * 2 + 512 * 4;
  node_msg_kernel<<<ZN / NODES_PER_BLK, BLK_MSG, smem_node, stream>>>(
      V0h, V0, edges, nbrs, nmask, nmW, nmb0, nmb1, nmb2, nmlng, nmlnb, V1, V1h);

  // kernel 2b: FFN (WMMA) -> outV, V2h
  size_t smem_ffn = (FW_HALFS + (size_t)FFN_WAVES * 16 * Hf) * 2;  // 320 KB
  ffn_kernel<<<ZN / FFN_ROWS, FFN_BLK, smem_ffn, stream>>>(
      V1h, V1, ffW, ffb0, ffb1, fflng, fflnb, outV, V2h);

  // kernel 3: edge messenger + LN -> outE
  size_t smem_edge = (W_HALFS + 2 * H_HALFS) * 2 + H_HALFS * 4;
  edge_msg_kernel<<<ZN / NODES_PER_BLK, BLK_MSG, smem_edge, stream>>>(
      V2h, edges, nbrs, nmask, emW, emb0, emb1, emb2, emlng, emlnb, outE);
}
